// DotProductAttention_65901978190341
// MI455X (gfx1250) — matrix operand order repaired and checked by reading, not compiled, not run on hardware
//
#include <hip/hip_runtime.h>

// ---------------- problem constants (from reference) ----------------
constexpr int kB = 4, kH = 16, kS = 2048, kD = 128;
constexpr int kQT    = 32;           // query rows per workgroup (2 x 16-row WMMA tiles)
constexpr int kWaves = 8;            // wave32s per workgroup
constexpr int kBlock = kWaves * 32;  // 256 threads
constexpr float kScale = 1.0f / 128.0f;  // reference divides by dk, not sqrt(dk)

typedef __attribute__((ext_vector_type(16))) __bf16 v16bf;
typedef __attribute__((ext_vector_type(16))) float  v16f;
typedef __attribute__((ext_vector_type(8)))  float  v8f;
typedef __attribute__((ext_vector_type(4)))  float  v4f;
typedef __attribute__((ext_vector_type(4)))  int    v4i;

// Address-space-qualified pointee typedefs.
typedef __attribute__((address_space(1))) v4i gv4i;  // global
typedef __attribute__((address_space(3))) v4i lv4i;  // LDS

#if __has_builtin(__builtin_amdgcn_global_load_async_to_lds_b128)
#define ASYNC_LDS 1
#else
#define ASYNC_LDS 0
#endif

__device__ __forceinline__ void wait_async_zero() {
#if ASYNC_LDS
#if __has_builtin(__builtin_amdgcn_s_wait_asynccnt)
  __builtin_amdgcn_s_wait_asynccnt(0);
#else
  asm volatile("s_wait_asynccnt 0x0" ::: "memory");
#endif
#endif
}

// Packed f32 -> bf16 conversion (v_cvt_pk_bf16_f32).
__device__ __forceinline__ v16bf cvt16bf(v16f f) {
  return __builtin_convertvector(f, v16bf);
}

// One workgroup: 32 query rows x full 2048-key strip for one (b,h).
// LDS: pbuf = 32x2048 f32 scores/probs (256KB)  -- needs CDNA5's 320KB WGP LDS,
//      vbuf = 2 x (32x128) f32 double-buffered V stage (32KB),
//      red  = 32x8 f32 softmax reduction scratch.
__global__ __launch_bounds__(kBlock)
void attn_fused_kernel(const float* __restrict__ Q, const float* __restrict__ K,
                       const float* __restrict__ V, const float* __restrict__ M,
                       float* __restrict__ out, float* __restrict__ attn) {
  extern __shared__ char smem_raw[];
  float* pbuf = reinterpret_cast<float*>(smem_raw);  // kQT * kS
  float* vbuf = pbuf + kQT * kS;                     // 2 * 32 * kD
  float* red  = vbuf + 2 * 32 * kD;                  // kQT * 8

  const int qt = blockIdx.x;       // 0..63
  const int bh = blockIdx.y;       // 0..63
  const int b  = bh / kH;
  const int q0 = qt * kQT;

  const float* Qb = Q + ((size_t)bh * kS + q0) * kD;
  const float* Kb = K + (size_t)bh * kS * kD;
  const float* Vb = V + (size_t)bh * kS * kD;
  const float* Mb = M + ((size_t)b * kS + q0) * kS;   // mask broadcast over heads
  float* Ob = out  + ((size_t)bh * kS + q0) * kD;
  float* Ab = attn + ((size_t)bh * kS + q0) * kS;

  const int tid  = threadIdx.x;
  const int wave = tid >> 5;
  const int lane = tid & 31;
  const int half = lane >> 4;   // which 16-lane half
  const int l16  = lane & 15;

  // ---- Q tile: 2 M-subtiles x 4 K-chunks of 16x32 bf16 A-fragments, pre-scaled ----
  // A layout (ISA 7.12.2): lane holds row m=l16; element e -> K = 16*(e/8) + 8*half + e%8
  v16bf qf[2][4];
#pragma unroll
  for (int mt = 0; mt < 2; ++mt) {
#pragma unroll
    for (int c = 0; c < 4; ++c) {
      v16f t;
#pragma unroll
      for (int e = 0; e < 16; ++e) {
        const int d = c * 32 + ((e >> 3) << 4) + (half << 3) + (e & 7);
        t[e] = Qb[(mt * 16 + l16) * kD + d] * kScale;
      }
      qf[mt][c] = cvt16bf(t);
    }
  }

  // ---- Phase 1: scores = (Q/dk) @ K^T  ->  pbuf ----
  // Each K B-fragment is reused for both query sub-tiles (2 WMMAs per fragment),
  // and the two accumulator chains interleave to hide WMMA->WMMA hazards.
  for (int t = wave; t < kS / 16; t += kWaves) {
    const int k0 = t * 16;
    const int kn = (k0 + 16 * kWaves < kS) ? (k0 + 16 * kWaves) : k0;
    __builtin_prefetch(&Kb[(size_t)(kn + l16) * kD], 0, 1);  // global_prefetch_b8

    v8f acc[2];
#pragma unroll
    for (int mt = 0; mt < 2; ++mt)
#pragma unroll
      for (int i = 0; i < 8; ++i) acc[mt][i] = 0.0f;

#pragma unroll
    for (int c = 0; c < 4; ++c) {
      // B fragment = K^T chunk: N = key l16, matmul-K = d-dim 16*(e/8) + 8*half + e%8
      v16f tf;
#pragma unroll
      for (int e = 0; e < 16; ++e) {
        const int d = c * 32 + ((e >> 3) << 4) + (half << 3) + (e & 7);
        tf[e] = Kb[(size_t)(k0 + l16) * kD + d];
      }
      const v16bf bf = cvt16bf(tf);
      acc[0] = __builtin_amdgcn_wmma_f32_16x16x32_bf16(false, qf[0][c], false, bf,
                                                       (short)0, acc[0], false, false);
      acc[1] = __builtin_amdgcn_wmma_f32_16x16x32_bf16(false, qf[1][c], false, bf,
                                                       (short)0, acc[1], false, false);
    }
    // C/D layout: VGPR r -> M = r + 8*half, N = l16
#pragma unroll
    for (int mt = 0; mt < 2; ++mt)
#pragma unroll
      for (int r = 0; r < 8; ++r)
        pbuf[(mt * 16 + r + (half << 3)) * kS + k0 + l16] = acc[mt][r];
  }
  __syncthreads();

  // ---- Phase 2: masked softmax per row (8 threads per row), write attn ----
  {
    const int row  = tid >> 3;        // 0..31
    const int part = tid & 7;
    const int kb   = part * (kS / 8); // 256 keys per part
    float* prow       = pbuf + row * kS;
    const float* mrow = Mb + (size_t)row * kS;
    float* arow       = Ab + (size_t)row * kS;

    float lmax = -3.0e38f;
    for (int k = kb; k < kb + kS / 8; ++k) {
      const float s = prow[k] + mrow[k] * (-1.0e9f);
      prow[k] = s;
      lmax = fmaxf(lmax, s);
    }
    red[row * 8 + part] = lmax;
    __syncthreads();
    float rmax = -3.0e38f;
#pragma unroll
    for (int i = 0; i < 8; ++i) rmax = fmaxf(rmax, red[row * 8 + i]);
    __syncthreads();

    float lsum = 0.0f;
    for (int k = kb; k < kb + kS / 8; ++k) {
      const float p = __expf(prow[k] - rmax);
      prow[k] = p;
      lsum += p;
    }
    red[row * 8 + part] = lsum;
    __syncthreads();
    float rsum = 0.0f;
#pragma unroll
    for (int i = 0; i < 8; ++i) rsum += red[row * 8 + i];
    const float inv = 1.0f / rsum;

    // Normalize in LDS, stream attn to HBM with non-temporal stores (never re-read).
    for (int k = kb; k < kb + kS / 8; k += 4) {
      v4f p4 = *reinterpret_cast<v4f*>(&prow[k]);
      p4 = p4 * inv;
      *reinterpret_cast<v4f*>(&prow[k]) = p4;
      __builtin_nontemporal_store(p4, reinterpret_cast<v4f*>(&arow[k]));
    }
  }
  __syncthreads();

  // ---- Phase 3: out = P @ V, double-buffered async V staging through LDS ----
  const int n0 = wave * 16;  // each wave owns a 16-wide slice of D
  v8f oacc[2];
#pragma unroll
  for (int mt = 0; mt < 2; ++mt)
#pragma unroll
    for (int i = 0; i < 8; ++i) oacc[mt][i] = 0.0f;

  auto stage = [&](int kc, int buf) {
    const char* gsrc = reinterpret_cast<const char*>(Vb + (size_t)kc * 32 * kD);
    char* ldst = reinterpret_cast<char*>(vbuf + buf * 32 * kD);
#if ASYNC_LDS
    // CDNA5 async memory->LDS copy, tracked with ASYNCcnt (no VGPR round trip).
    for (int i = tid; i < (32 * kD * 4) / 16; i += kBlock) {
      __builtin_amdgcn_global_load_async_to_lds_b128(
          (gv4i*)(gsrc + i * 16), (lv4i*)(ldst + i * 16), 0, 0);
    }
#else
    const v4f* src = reinterpret_cast<const v4f*>(gsrc);
    v4f* dst = reinterpret_cast<v4f*>(ldst);
    for (int i = tid; i < 32 * kD / 4; i += kBlock) dst[i] = src[i];
#endif
  };

  stage(0, 0);
  wait_async_zero();
  __syncthreads();

  constexpr int kChunks = kS / 32;  // 64
  for (int kc = 0; kc < kChunks; ++kc) {
    const int cur = kc & 1;
    if (kc + 1 < kChunks) stage(kc + 1, cur ^ 1);  // overlap next V chunk with WMMA

    const int k0 = kc * 32;
    const float* vb = vbuf + cur * 32 * kD;

    // B fragment (expensive strided column gather) built once, used for 2 WMMAs.
    v16f tb;
#pragma unroll
    for (int e = 0; e < 16; ++e) {   // B: V column n0+l16, K = 16*(e/8) + 8*half + e%8
      const int kk = ((e >> 3) << 4) + (half << 3) + (e & 7);
      tb[e] = vb[kk * kD + n0 + l16];
    }
    const v16bf bfv = cvt16bf(tb);

#pragma unroll
    for (int mt = 0; mt < 2; ++mt) {
      v16f ta;
#pragma unroll
      for (int e = 0; e < 16; ++e) {   // A: probs rows mt*16 + l16
        const int kk = ((e >> 3) << 4) + (half << 3) + (e & 7);
        ta[e] = pbuf[(mt * 16 + l16) * kS + k0 + kk];
      }
      oacc[mt] = __builtin_amdgcn_wmma_f32_16x16x32_bf16(false, cvt16bf(ta), false, bfv,
                                                         (short)0, oacc[mt], false, false);
    }

    wait_async_zero();   // this wave's async loads for the next buffer have landed
    __syncthreads();     // everyone done reading cur before it is overwritten next iter
  }
#pragma unroll
  for (int mt = 0; mt < 2; ++mt)
#pragma unroll
    for (int r = 0; r < 8; ++r)
      __builtin_nontemporal_store(
          oacc[mt][r], &Ob[(size_t)(mt * 16 + r + (half << 3)) * kD + n0 + l16]);
}

extern "C" void kernel_launch(void* const* d_in, const int* in_sizes, int n_in,
                              void* d_out, int out_size, void* d_ws, size_t ws_size,
                              hipStream_t stream) {
  const float* Q = (const float*)d_in[0];
  const float* K = (const float*)d_in[1];
  const float* V = (const float*)d_in[2];
  const float* M = (const float*)d_in[3];
  float* out  = (float*)d_out;
  float* attn = out + (size_t)kB * kH * kS * kD;  // tuple order: (out, attn)

  constexpr size_t shmem =
      (size_t)(kQT * kS + 2 * 32 * kD + kQT * 8) * sizeof(float);  // ~290 KB
  static_assert(shmem <= 320u * 1024u, "fits CDNA5 WGP LDS");

  (void)hipFuncSetAttribute(reinterpret_cast<const void*>(&attn_fused_kernel),
                            hipFuncAttributeMaxDynamicSharedMemorySize, (int)shmem);

  dim3 grid(kS / kQT, kB * kH);  // 64 x 64 = 4096 workgroups
  attn_fused_kernel<<<grid, kBlock, shmem, stream>>>(Q, K, V, M, out, attn);

  (void)in_sizes; (void)n_in; (void)out_size; (void)d_ws; (void)ws_size;
}
